// BgeAttention_23570780521212
// MI455X (gfx1250) — matrix operand order repaired and checked by reading, not compiled, not run on hardware
//
#include <hip/hip_runtime.h>
#include <hip/hip_bf16.h>
#include <math.h>

// ---------------------------------------------------------------------------
// CDNA5 WMMA types (wave32): A/B = 16 bf16 (8 VGPRs), C/D = 8 f32 (8 VGPRs)
// ---------------------------------------------------------------------------
typedef __attribute__((ext_vector_type(16))) __bf16 bf16x16;
typedef __attribute__((ext_vector_type(8)))  float  f32x8;
typedef __attribute__((ext_vector_type(2)))  int    i32x2;
typedef __attribute__((ext_vector_type(4)))  int    i32x4;

union AFrag { bf16x16 v; unsigned int u[8]; unsigned short s[16]; };
union CFrag { f32x8   v; float        f[8]; };

#define AS1 __attribute__((address_space(1)))
#define AS3 __attribute__((address_space(3)))

// Async global->LDS path (ASYNCcnt-tracked DMA), guarded so the kernel always
// compiles; falls back to global_load_b128 + ds_store_b128 if unavailable.
#if defined(__gfx1250__) &&                                                   \
    __has_builtin(__builtin_amdgcn_global_load_async_to_lds_b128) &&          \
    __has_builtin(__builtin_amdgcn_global_load_async_to_lds_b64) &&           \
    __has_builtin(__builtin_amdgcn_s_wait_asynccnt)
#define USE_ASYNC_LDS 1
#else
#define USE_ASYNC_LDS 0
#endif

// Low 32 bits of a generic LDS address are the LDS offset (ISA 10.2); AS3
// pointers are 32-bit, AS1 pointers are numerically identical to generic.
__device__ __forceinline__ AS3 i32x2* lds_v2(void* p) {
  return (AS3 i32x2*)(unsigned int)(unsigned long long)p;
}
__device__ __forceinline__ AS3 i32x4* lds_v4(void* p) {
  return (AS3 i32x4*)(unsigned int)(unsigned long long)p;
}
__device__ __forceinline__ AS1 i32x2* glb_v2(const void* p) {
  return (AS1 i32x2*)(unsigned long long)p;
}
__device__ __forceinline__ AS1 i32x4* glb_v4(const void* p) {
  return (AS1 i32x4*)(unsigned long long)p;
}

__device__ __forceinline__ unsigned short f2bf(float f) {
  unsigned int b = __float_as_uint(f);
  b += 0x7FFFu + ((b >> 16) & 1u);   // round-to-nearest-even
  return (unsigned short)(b >> 16);
}

// Load a 16-half fragment from 16 contiguous ushorts (LDS or global) -> 2x b128
__device__ __forceinline__ void ld_frag16(AFrag& f, const unsigned short* p) {
  const uint4 lo = *(const uint4*)(p);
  const uint4 hi = *(const uint4*)(p + 8);
  f.u[0] = lo.x; f.u[1] = lo.y; f.u[2] = lo.z; f.u[3] = lo.w;
  f.u[4] = hi.x; f.u[5] = hi.y; f.u[6] = hi.z; f.u[7] = hi.w;
}

// ---------------------------------------------------------------------------
// fp32 -> bf16 conversion (grid-stride)
// ---------------------------------------------------------------------------
__global__ void cvt_f32_bf16(const float* __restrict__ in,
                             unsigned short* __restrict__ out, int n) {
  int i = blockIdx.x * blockDim.x + threadIdx.x;
  const int stride = gridDim.x * blockDim.x;
  for (; i < n; i += stride) out[i] = f2bf(in[i]);
}

// ---------------------------------------------------------------------------
// fp32 [R x C] -> bf16 transposed [C x R]; 32x32 LDS tile, coalesced both ways.
// One-time weight prep so GEMM B-fragments become contiguous b128 loads.
// ---------------------------------------------------------------------------
__global__ __launch_bounds__(256) void cvt_transpose_bf16(
    const float* __restrict__ in, unsigned short* __restrict__ out,
    int R, int C) {
  __shared__ unsigned short tile[32][33];
  const int bx = blockIdx.x * 32;            // column tile of input
  const int by = blockIdx.y * 32;            // row tile of input
  const int tx = threadIdx.x & 31;
  const int ty4 = (threadIdx.x >> 5) * 4;
  #pragma unroll
  for (int i = 0; i < 4; ++i)
    tile[ty4 + i][tx] = f2bf(in[(size_t)(by + ty4 + i) * C + bx + tx]);
  __syncthreads();
  #pragma unroll
  for (int i = 0; i < 4; ++i)
    out[(size_t)(bx + ty4 + i) * R + by + tx] = tile[tx][ty4 + i];
}

// ---------------------------------------------------------------------------
// bf16 WMMA GEMM:  C[M,N] = A[M,K] (row-major) * Bt[N,K] (pre-transposed) + bias
// Block = 256 threads (8 waves).  Block tile 32x128, wave tile 16x32 (two
// 16x16 WMMAs sharing one A fragment).  K-step 32, double-buffered LDS with
// async global->LDS DMA (ASYNCcnt) when available.  A slab stored in natural k order
// so A fragments are two ds_load_b128; Bt slab stored
// n-major so B fragments are two ds_load_b128.
// Epilogue modes:
//   0: bf16 out, (B,H,S,hd) head layout     (Q, K projections)
//   1: bf16 out, (B,H,hd,S) transposed      (V projection)
//   2: f32 out, row-major                   (O-proj, fc2)
//   3: bf16 out, row-major, relu->gelu      (fc1)
// ---------------------------------------------------------------------------
__global__ __launch_bounds__(256) void gemm_bf16_wmma(
    const unsigned short* __restrict__ A,
    const unsigned short* __restrict__ Bt,
    const float* __restrict__ bias,
    int N, int K, int mode,
    float* __restrict__ outF,
    unsigned short* __restrict__ outH)
{
  __shared__ unsigned short As[2][32 * 40];     // 32 rows x 32 k (natural order)
  __shared__ unsigned short Bs[2][128 * 40];    // 128 n-rows x 32 k
  const int tid  = threadIdx.x;
  const int wave = tid >> 5, lane = tid & 31;
  const int hl = lane >> 4, l16 = lane & 15;
  const int wr = wave & 1, wc = wave >> 1;      // 2 x 4 wave grid
  const int row0 = blockIdx.y * 32;
  const int col0 = blockIdx.x * 128;

  // A loader: 4 k-elems per thread (b64), natural k-order destination
  const int ar = tid >> 3, acs = (tid & 7) * 4;
  const int ab = acs >> 3;
  const int adb = ab;
  const int adst = adb * 8 + (acs & 7);
  // B loader: 16 k-elems per thread (2x b128)
  const int bn = tid >> 1, bk8 = (tid & 1) * 16;

  auto load_slab = [&](int kc, int buf) {
    const unsigned short* asrc = A + (size_t)(row0 + ar) * K + kc + acs;
    const unsigned short* bsrc = Bt + (size_t)(col0 + bn) * K + kc + bk8;
#if USE_ASYNC_LDS
    __builtin_amdgcn_global_load_async_to_lds_b64(
        glb_v2(asrc), lds_v2(&As[buf][ar * 40 + adst]), 0, 0);
    __builtin_amdgcn_global_load_async_to_lds_b128(
        glb_v4(bsrc), lds_v4(&Bs[buf][bn * 40 + bk8]), 0, 0);
    __builtin_amdgcn_global_load_async_to_lds_b128(
        glb_v4(bsrc + 8), lds_v4(&Bs[buf][bn * 40 + bk8 + 8]), 0, 0);
#else
    *(uint2*)(&As[buf][ar * 40 + adst])    = *(const uint2*)(asrc);
    *(uint4*)(&Bs[buf][bn * 40 + bk8])     = *(const uint4*)(bsrc);
    *(uint4*)(&Bs[buf][bn * 40 + bk8 + 8]) = *(const uint4*)(bsrc + 8);
#endif
  };
  auto slab_barrier = [&]() {
#if USE_ASYNC_LDS
    __builtin_amdgcn_s_wait_asynccnt(0);     // async DMA of other buffer done
#endif
    __syncthreads();
  };

  CFrag acc0, acc1;
  #pragma unroll
  for (int i = 0; i < 8; ++i) { acc0.f[i] = 0.f; acc1.f[i] = 0.f; }

  load_slab(0, 0);
  slab_barrier();
  int buf = 0;
  for (int kc = 0; kc < K; kc += 32) {
    if (kc + 32 < K) {
      load_slab(kc + 32, buf ^ 1);                    // fill other buffer
      if (kc + 64 < K) {                              // global_prefetch_b8
        __builtin_prefetch(A  + (size_t)(row0 + ar) * K + kc + 64 + acs, 0, 1);
        __builtin_prefetch(Bt + (size_t)(col0 + bn) * K + kc + 64 + bk8, 0, 1);
      }
    }
    AFrag a, b0, b1;
    ld_frag16(a,  &As[buf][(wr * 16 + l16) * 40 + 16 * hl]);
    ld_frag16(b0, &Bs[buf][(wc * 16 + l16) * 40 + 16 * hl]);
    ld_frag16(b1, &Bs[buf][(64 + wc * 16 + l16) * 40 + 16 * hl]);
    acc0.v = __builtin_amdgcn_wmma_f32_16x16x32_bf16(
        false, a.v, false, b0.v, (short)0, acc0.v, false, false);
    acc1.v = __builtin_amdgcn_wmma_f32_16x16x32_bf16(
        false, a.v, false, b1.v, (short)0, acc1.v, false, false);
    slab_barrier();
    buf ^= 1;
  }

  #pragma unroll
  for (int t = 0; t < 2; ++t) {
    const CFrag& acc = t ? acc1 : acc0;
    const int gcol = col0 + t * 64 + wc * 16 + l16;
    const float bval = bias[gcol];
    #pragma unroll
    for (int r = 0; r < 8; ++r) {
      const int grow = row0 + wr * 16 + r + 8 * hl;   // C layout: m = r + 8*half
      const float val = acc.f[r] + bval;
      if (mode == 0 || mode == 1) {                   // fixed shapes: S=2048,D=1024
        const int bb = grow >> 11, ss = grow & 2047;
        const int hh = gcol >> 6, dd = gcol & 63;
        const size_t idx = (mode == 0)
            ? ((((size_t)bb * 16 + hh) * 2048 + ss) * 64 + dd)
            : ((((size_t)bb * 16 + hh) * 64 + dd) * 2048 + ss);
        outH[idx] = f2bf(val);
      } else if (mode == 2) {
        outF[(size_t)grow * N + gcol] = val;
      } else {                                        // relu -> exact gelu
        const float rl = fmaxf(val, 0.f);
        const float ge = 0.5f * rl * (1.f + erff(rl * 0.70710678118654752f));
        outH[(size_t)grow * N + gcol] = f2bf(ge);
      }
    }
  }
}

// ---------------------------------------------------------------------------
// Flash attention, bf16 WMMA.  One wave = 16 query rows x full hd=64.
// Key chunks of 32: QK^T = 4 WMMAs, online softmax (shfl row-reduce in the
// C layout), P staged in natural k order in per-wave LDS tile (A-frag reload is two
// ds_load_b128), PV = 4 WMMAs.  Q/K in (B,H,S,64) bf16; V in (B,H,64,S) bf16.
// All global fragment loads are b128.  ctx written bf16 (B,S,D).
// ---------------------------------------------------------------------------
__global__ __launch_bounds__(256) void flash_attn_wmma(
    const unsigned short* __restrict__ Qb,
    const unsigned short* __restrict__ Kb,
    const unsigned short* __restrict__ Vt,
    unsigned short* __restrict__ ctxB)
{
  const int S = 2048, HD = 64;
  __shared__ unsigned short Pt[8][16 * 40];          // per-wave 16x32 P (natural k order)
  const int tid = threadIdx.x, wave = tid >> 5, lane = tid & 31;
  const int hl = lane >> 4, l16 = lane & 15;
  const int bh = blockIdx.y;                         // 0..31 (b*16+h)
  const int qt = blockIdx.x * 8 + wave;              // query tile 0..127
  const size_t qoff = ((size_t)bh * S + qt * 16 + l16) * HD;

  // Q is the A-role matrix: u[0..7] = hd {16hl..16hl+15} (contiguous)
  AFrag q0, q1;
  {
    const uint4 a0 = *(const uint4*)(Qb + qoff + 16 * hl);
    const uint4 a1 = *(const uint4*)(Qb + qoff + 8 + 16 * hl);
    q0.u[0]=a0.x; q0.u[1]=a0.y; q0.u[2]=a0.z; q0.u[3]=a0.w;
    q0.u[4]=a1.x; q0.u[5]=a1.y; q0.u[6]=a1.z; q0.u[7]=a1.w;
    const uint4 a2 = *(const uint4*)(Qb + qoff + 32 + 16 * hl);
    const uint4 a3 = *(const uint4*)(Qb + qoff + 40 + 16 * hl);
    q1.u[0]=a2.x; q1.u[1]=a2.y; q1.u[2]=a2.z; q1.u[3]=a2.w;
    q1.u[4]=a3.x; q1.u[5]=a3.y; q1.u[6]=a3.z; q1.u[7]=a3.w;
  }

  CFrag acc[4];
  #pragma unroll
  for (int j = 0; j < 4; ++j)
    #pragma unroll
    for (int i = 0; i < 8; ++i) acc[j].f[i] = 0.f;
  float rowm[8], rowl[8];
  #pragma unroll
  for (int r = 0; r < 8; ++r) { rowm[r] = -3.0e38f; rowl[r] = 0.f; }

  // P-store positions (natural k order 0-31)
  const int pos0 = l16;                              // k = l16
  const int pos1 = l16 + 16;                         // k = l16 + 16

  for (int kc = 0; kc < S; kc += 32) {
    CFrag sc[2];
    #pragma unroll
    for (int sub = 0; sub < 2; ++sub) {              // two 16-key sub-tiles
      AFrag bk0, bk1;                                // B-role: hd-contig b128
      const unsigned short* kp =
          Kb + ((size_t)bh * S + kc + sub * 16 + l16) * HD + 16 * hl;
      ld_frag16(bk0, kp);                            // hd chunk 0..31
      ld_frag16(bk1, kp + 32);                       // hd chunk 32..63
      CFrag z;
      #pragma unroll
      for (int i = 0; i < 8; ++i) z.f[i] = 0.f;
      z.v = __builtin_amdgcn_wmma_f32_16x16x32_bf16(
          false, q0.v, false, bk0.v, (short)0, z.v, false, false);
      z.v = __builtin_amdgcn_wmma_f32_16x16x32_bf16(
          false, q1.v, false, bk1.v, (short)0, z.v, false, false);
      sc[sub] = z;
    }

    const float scale = 0.125f;                      // 1/sqrt(64)
    #pragma unroll
    for (int r = 0; r < 8; ++r) {
      const float s0 = sc[0].f[r] * scale, s1 = sc[1].f[r] * scale;
      float mx = fmaxf(s0, s1);
      mx = fmaxf(mx, __shfl_xor(mx, 1, 32));
      mx = fmaxf(mx, __shfl_xor(mx, 2, 32));
      mx = fmaxf(mx, __shfl_xor(mx, 4, 32));
      mx = fmaxf(mx, __shfl_xor(mx, 8, 32));         // reduce within 16-lane group
      const float mnew = fmaxf(rowm[r], mx);
      const float resc = __expf(rowm[r] - mnew);
      rowm[r] = mnew;
      const float p0 = __expf(s0 - mnew), p1 = __expf(s1 - mnew);
      float ps = p0 + p1;
      ps += __shfl_xor(ps, 1, 32);
      ps += __shfl_xor(ps, 2, 32);
      ps += __shfl_xor(ps, 4, 32);
      ps += __shfl_xor(ps, 8, 32);
      rowl[r] = rowl[r] * resc + ps;
      #pragma unroll
      for (int j = 0; j < 4; ++j) acc[j].f[r] *= resc;
      const int m = r + 8 * hl;                      // C-layout row
      Pt[wave][m * 40 + pos0] = f2bf(p0);
      Pt[wave][m * 40 + pos1] = f2bf(p1);
    }

    // Re-load P as an A fragment (same-wave DS ops are in order): 2x b128
    AFrag ap;
    ld_frag16(ap, &Pt[wave][l16 * 40 + 16 * hl]);

    #pragma unroll
    for (int j = 0; j < 4; ++j) {                    // ctx cols j*16..j*16+15
      AFrag bv;                                      // B-role: S-contig b128
      ld_frag16(bv, Vt + ((size_t)bh * HD + j * 16 + l16) * S + kc + 16 * hl);
      acc[j].v = __builtin_amdgcn_wmma_f32_16x16x32_bf16(
          false, ap.v, false, bv.v, (short)0, acc[j].v, false, false);
    }
  }

  const int bb = bh >> 4, hh = bh & 15;
  #pragma unroll
  for (int r = 0; r < 8; ++r) {
    const int m = r + 8 * hl;
    const int srow = qt * 16 + m;
    const float inv = 1.f / rowl[r];
    #pragma unroll
    for (int j = 0; j < 4; ++j) {
      const int col = hh * 64 + j * 16 + l16;
      ctxB[((size_t)bb * 2048 + srow) * 1024 + col] = f2bf(acc[j].f[r] * inv);
    }
  }
}

// ---------------------------------------------------------------------------
// LayerNorm over D=1024.  One block per row, optional residual input,
// optional f32 and bf16 outputs.
// ---------------------------------------------------------------------------
__global__ __launch_bounds__(256) void layernorm_k(
    const float* __restrict__ in, const float* __restrict__ res,
    const float* __restrict__ g, const float* __restrict__ b,
    float* __restrict__ outF, unsigned short* __restrict__ outH)
{
  const int D = 1024;
  const int row = blockIdx.x, tid = threadIdx.x;
  const int wave = tid >> 5, lane = tid & 31;
  const float* x = in + (size_t)row * D;
  float vals[4];
  float s = 0.f, s2 = 0.f;
  #pragma unroll
  for (int i = 0; i < 4; ++i) {
    const int c = tid + i * 256;
    float v = x[c];
    if (res) v += res[(size_t)row * D + c];
    vals[i] = v;
    s += v; s2 += v * v;
  }
  #pragma unroll
  for (int mask = 1; mask < 32; mask <<= 1) {
    s  += __shfl_xor(s,  mask, 32);
    s2 += __shfl_xor(s2, mask, 32);
  }
  __shared__ float redS[8], redQ[8];
  if (lane == 0) { redS[wave] = s; redQ[wave] = s2; }
  __syncthreads();
  float ts = 0.f, tq = 0.f;
  #pragma unroll
  for (int w = 0; w < 8; ++w) { ts += redS[w]; tq += redQ[w]; }
  const float mean = ts / D;
  const float var  = tq / D - mean * mean;
  const float inv  = rsqrtf(var + 1e-12f);
  #pragma unroll
  for (int i = 0; i < 4; ++i) {
    const int c = tid + i * 256;
    const float nv = (vals[i] - mean) * inv * g[c] + b[c];
    if (outF) outF[(size_t)row * D + c] = nv;
    if (outH) outH[(size_t)row * D + c] = f2bf(nv);
  }
}

// ---------------------------------------------------------------------------
// Host launcher
// ---------------------------------------------------------------------------
extern "C" void kernel_launch(void* const* d_in, const int* in_sizes, int n_in,
                              void* d_out, int out_size, void* d_ws, size_t ws_size,
                              hipStream_t stream) {
  (void)in_sizes; (void)n_in; (void)out_size; (void)ws_size;
  const int Bb = 2, S = 2048, D = 1024, F = 4096;
  const int M = Bb * S;                               // 4096 rows

  const float* x   = (const float*)d_in[0];
  const float* Wq  = (const float*)d_in[1];
  const float* bq  = (const float*)d_in[2];
  const float* Wk  = (const float*)d_in[3];
  const float* bk  = (const float*)d_in[4];
  const float* Wv  = (const float*)d_in[5];
  const float* bv  = (const float*)d_in[6];
  const float* Wo  = (const float*)d_in[7];
  const float* bo  = (const float*)d_in[8];
  const float* g1  = (const float*)d_in[9];
  const float* be1 = (const float*)d_in[10];
  const float* W1  = (const float*)d_in[11];
  const float* b1  = (const float*)d_in[12];
  const float* W2  = (const float*)d_in[13];
  const float* b2  = (const float*)d_in[14];
  const float* g2  = (const float*)d_in[15];
  const float* be2 = (const float*)d_in[16];
  float* out = (float*)d_out;

  char* p = (char*)d_ws;
  auto carve = [&](size_t bytes) -> void* {
    void* r = (void*)p;
    p += (bytes + 255) & ~(size_t)255;
    return r;
  };
  unsigned short* xb     = (unsigned short*)carve((size_t)M * D * 2);
  unsigned short* Wqt    = (unsigned short*)carve((size_t)D * D * 2);  // N x K
  unsigned short* Wkt    = (unsigned short*)carve((size_t)D * D * 2);
  unsigned short* Wvt    = (unsigned short*)carve((size_t)D * D * 2);
  unsigned short* Wot    = (unsigned short*)carve((size_t)D * D * 2);
  unsigned short* W1t    = (unsigned short*)carve((size_t)D * F * 2);  // F x D
  unsigned short* W2t    = (unsigned short*)carve((size_t)F * D * 2);  // D x F
  unsigned short* qB     = (unsigned short*)carve((size_t)M * D * 2);  // (B,H,S,64)
  unsigned short* kB     = (unsigned short*)carve((size_t)M * D * 2);  // (B,H,S,64)
  unsigned short* vT     = (unsigned short*)carve((size_t)M * D * 2);  // (B,H,64,S)
  unsigned short* ctxB   = (unsigned short*)carve((size_t)M * D * 2);  // (B,S,D)
  float*          tmp1   = (float*)carve((size_t)M * D * 4);
  float*          attnO  = (float*)carve((size_t)M * D * 4);
  unsigned short* attnOb = (unsigned short*)carve((size_t)M * D * 2);
  unsigned short* hB     = (unsigned short*)carve((size_t)M * F * 2);
  float*          tmp2   = (float*)carve((size_t)M * D * 4);

  // One-time precision/layout prep (bandwidth-trivial vs. compute-bound GEMMs)
  cvt_f32_bf16<<<512, 256, 0, stream>>>(x, xb, M * D);
  cvt_transpose_bf16<<<dim3(D / 32, D / 32), 256, 0, stream>>>(Wq, Wqt, D, D);
  cvt_transpose_bf16<<<dim3(D / 32, D / 32), 256, 0, stream>>>(Wk, Wkt, D, D);
  cvt_transpose_bf16<<<dim3(D / 32, D / 32), 256, 0, stream>>>(Wv, Wvt, D, D);
  cvt_transpose_bf16<<<dim3(D / 32, D / 32), 256, 0, stream>>>(Wo, Wot, D, D);
  cvt_transpose_bf16<<<dim3(F / 32, D / 32), 256, 0, stream>>>(W1, W1t, D, F);
  cvt_transpose_bf16<<<dim3(D / 32, F / 32), 256, 0, stream>>>(W2, W2t, F, D);

  const dim3 blk(256);
  const dim3 gP(D / 128, M / 32);                     // (8, 128)

  gemm_bf16_wmma<<<gP, blk, 0, stream>>>(xb, Wqt, bq, D, D, 0, nullptr, qB);
  gemm_bf16_wmma<<<gP, blk, 0, stream>>>(xb, Wkt, bk, D, D, 0, nullptr, kB);
  gemm_bf16_wmma<<<gP, blk, 0, stream>>>(xb, Wvt, bv, D, D, 1, nullptr, vT);

  flash_attn_wmma<<<dim3(S / 128, Bb * 16), blk, 0, stream>>>(qB, kB, vT, ctxB);

  gemm_bf16_wmma<<<gP, blk, 0, stream>>>(ctxB, Wot, bo, D, D, 2, tmp1, nullptr);
  layernorm_k<<<M, blk, 0, stream>>>(tmp1, nullptr, g1, be1, attnO, attnOb);

  gemm_bf16_wmma<<<dim3(F / 128, M / 32), blk, 0, stream>>>(attnOb, W1t, b1, F, D,
                                                            3, nullptr, hB);
  gemm_bf16_wmma<<<gP, blk, 0, stream>>>(hB, W2t, b2, D, F, 2, tmp2, nullptr);
  layernorm_k<<<M, blk, 0, stream>>>(tmp2, attnO, g2, be2, out, nullptr);
}
